// GraphAttentionLayer_36189394436297
// MI455X (gfx1250) — matrix operand order repaired and checked by reading, not compiled, not run on hardware
//
#include <hip/hip_runtime.h>
#include <hip/hip_bf16.h>

typedef __attribute__((ext_vector_type(16))) __bf16 v16bf;
typedef __attribute__((ext_vector_type(8)))  float  v8f;

#define N_NODES 4096
#define IN_FEAT 128
#define OUT_FEAT 64
#define N_HEADS 4
#define CDIM (OUT_FEAT * N_HEADS) /* 256 */
#define N_SEG 4
#define SEG_COLS (N_NODES / N_SEG) /* 1024 */
#define LRELU_ALPHA 0.2f
#define LOG2E 1.4426950408889634f
#define NEG_FILL_L2 (-9.0e15f * LOG2E)   /* mask fill, already in log2 domain */

__device__ __forceinline__ float lane_bcast(float v, int srclane) {
    // wave32: ds_bpermute gathers from lane (addr>>2) & 31
    return __builtin_bit_cast(float,
        __builtin_amdgcn_ds_bpermute(srclane << 2, __builtin_bit_cast(int, v)));
}

// ---------------------------------------------------------------------------
// Kernel 1: Wh = h @ W  (fp32, 4096x128 @ 128x256).  Also emit WhT in bf16,
// transposed [feat][node], so the attention kernel's WMMA B-fragments are
// contiguous 32B loads per lane.
// ---------------------------------------------------------------------------
__global__ __launch_bounds__(256) void k_gemm_wh(const float* __restrict__ h,
                                                 const float* __restrict__ W,
                                                 float* __restrict__ wh,
                                                 __bf16* __restrict__ whT) {
    int idx = blockIdx.x * 256 + threadIdx.x;   // idx = n*256 + c
    int n = idx >> 8;
    int c = idx & 255;
    const float* hr = h + (size_t)n * IN_FEAT;  // uniform per block -> scalar loads
    float s = 0.f;
#pragma unroll 8
    for (int k = 0; k < IN_FEAT; ++k)
        s = fmaf(hr[k], W[(size_t)k * CDIM + c], s);
    wh[idx] = s;
    whT[(size_t)c * N_NODES + n] = (__bf16)s;
}

// ---------------------------------------------------------------------------
// Kernel 2: left/right attention projections, stored transposed [head][node],
// pre-scaled by log2(e) so the attention kernel works purely in exp2 domain
// (LeakyReLU commutes with multiplication by a positive constant).
// ---------------------------------------------------------------------------
__global__ __launch_bounds__(256) void k_leftright(const float* __restrict__ wh,
                                                   const float* __restrict__ a,
                                                   float* __restrict__ leftT,
                                                   float* __restrict__ rightT) {
    int idx = blockIdx.x * 256 + threadIdx.x;   // idx = head*4096 + n
    int head = idx >> 12;
    int n = idx & (N_NODES - 1);
    const float* whr = wh + (size_t)n * CDIM + head * OUT_FEAT;
    float L = 0.f, R = 0.f;
#pragma unroll
    for (int f = 0; f < OUT_FEAT; ++f) {
        float v = whr[f];
        L = fmaf(v, a[f], L);
        R = fmaf(v, a[OUT_FEAT + f], R);
    }
    leftT[idx]  = L * LOG2E;
    rightT[idx] = R * LOG2E;
}

// ---------------------------------------------------------------------------
// Kernel 3: flash-style masked softmax + attn @ Wh via bf16 WMMA.
// Block = 512 threads = 16 waves: wave (head, seg) scans its 1024-column
// segment for one 16-row tile.  Partial (m, l, acc) combined via LDS.
// head/seg are readfirstlane'd so loop bounds & bases live in SGPRs (no
// exec-masked loop; WMMA sees EXEC == all-ones trivially).
//
// A-fragment (16x32 bf16) lane layout (K order chosen here, same as B): lane l holds row M=l&15;
// element e maps to K = 16*(l>>4) + e  within the 32-chunk, i.e.
// two contiguous 8-wide K runs -> adj fetched as 4x int4, right as 4x float4.
// B-fragment (32x16): lane l holds col N=l&15, K = 16*(l>>4)+e -> contiguous
// 16 bf16 from WhT = one 32-byte load.
// C/D: VGPR r holds row M = r + 8*(l>>4), col N = l&15.
// ---------------------------------------------------------------------------
__global__ __launch_bounds__(512) void k_gat_attn(const int* __restrict__ adj,
                                                  const float* __restrict__ leftT,
                                                  const float* __restrict__ rightT,
                                                  const __bf16* __restrict__ whT,
                                                  const float* __restrict__ bias,
                                                  float* __restrict__ out) {
    extern __shared__ float smem[];
    float* ls_acc = smem;                       // [head][seg][row16][feat64] = 16384 f
    float* ls_m   = smem + N_HEADS * N_SEG * 16 * OUT_FEAT;  // [head][seg][row16] = 256 f
    float* ls_l   = ls_m + N_HEADS * N_SEG * 16;             // 256 f

    // wave id is uniform within a wave: force it scalar
    const int wave = __builtin_amdgcn_readfirstlane((int)threadIdx.x >> 5);  // 0..15
    const int head = wave >> 2;          // 0..3 (SGPR)
    const int seg  = wave & 3;           // 0..3 (SGPR)
    const int lane = threadIdx.x & 31;
    const int m    = lane & 15;
    const int hi   = lane >> 4;
    const int row0 = blockIdx.x * 16;
    const int row  = row0 + m;

    const float leftv = leftT[head * N_NODES + row];
    const int   adj_off = row * N_NODES;                       // lane offset into adj
    const float* rrow   = rightT + head * N_NODES;             // scalar base
    const int   b_off   = (head * OUT_FEAT + m) * N_NODES + 16 * hi;

    v8f acc[4];
#pragma unroll
    for (int t = 0; t < 4; ++t) acc[t] = (v8f){0.f,0.f,0.f,0.f,0.f,0.f,0.f,0.f};

    float mrun = -3.0e38f;  // running max (log2 domain; NEG_FILL_L2 > this)
    float lrun = 0.f;       // running denominator

    const int cbeg = seg * SEG_COLS;                           // SGPR
    const int cend = cbeg + SEG_COLS;                          // SGPR
    for (int c0 = cbeg; c0 < cend; c0 += 32) {                 // scalar loop
        const int kb0 = c0 + 16 * hi;
        const int kb1 = c0 + 16 * hi + 8;

        int4 ad[4];
        ad[0] = *(const int4*)(adj + adj_off + kb0);
        ad[1] = *(const int4*)(adj + adj_off + kb0 + 4);
        ad[2] = *(const int4*)(adj + adj_off + kb1);
        ad[3] = *(const int4*)(adj + adj_off + kb1 + 4);
        float4 rv[4];
        rv[0] = *(const float4*)(rrow + kb0);
        rv[1] = *(const float4*)(rrow + kb0 + 4);
        rv[2] = *(const float4*)(rrow + kb1);
        rv[3] = *(const float4*)(rrow + kb1 + 4);

        float e[16];
#pragma unroll
        for (int q = 0; q < 4; ++q) {
            const int*   ai = (const int*)&ad[q];
            const float* ri = (const float*)&rv[q];
#pragma unroll
            for (int j = 0; j < 4; ++j) {
                float ev = leftv + ri[j];                       // log2-domain logit
                ev = (ev >= 0.f) ? ev : LRELU_ALPHA * ev;       // LeakyReLU
                e[q * 4 + j] = (ai[j] > 0) ? ev : NEG_FILL_L2;  // adjacency mask
            }
        }

        // chunk max over this lane's 16 cols, then across the lane pair (l, l^16)
        float cm = e[0];
#pragma unroll
        for (int i = 1; i < 16; ++i) cm = fmaxf(cm, e[i]);
        cm = fmaxf(cm, lane_bcast(cm, lane ^ 16));
        float mnew = fmaxf(mrun, cm);

        v16bf afrag;
        float s = 0.f;
#pragma unroll
        for (int i = 0; i < 16; ++i) {
            float p = __builtin_amdgcn_exp2f(e[i] - mnew);
            s += p;
            afrag[i] = (__bf16)p;   // element order == K order (see layout note)
        }
        s += lane_bcast(s, lane ^ 16);

        // rescale only when some row's max actually moved (wave-uniform branch,
        // EXEC stays all-ones inside -> WMMA-safe)
        if (__builtin_amdgcn_ballot_w32(mnew > mrun)) {
            float alpha = __builtin_amdgcn_exp2f(mrun - mnew);
            lrun = lrun * alpha + s;
            float al[8];
#pragma unroll
            for (int r = 0; r < 8; ++r) al[r] = lane_bcast(alpha, 8 * hi + r);
#pragma unroll
            for (int t = 0; t < 4; ++t)
#pragma unroll
                for (int r = 0; r < 8; ++r) acc[t][r] *= al[r];
        } else {
            lrun += s;
        }
        mrun = mnew;

        // B fragments: contiguous 32B per lane from transposed bf16 Wh
        const int bo = b_off + c0;
        v16bf b0 = *(const v16bf*)(whT + bo);
        v16bf b1 = *(const v16bf*)(whT + bo + 16 * N_NODES);
        v16bf b2 = *(const v16bf*)(whT + bo + 32 * N_NODES);
        v16bf b3 = *(const v16bf*)(whT + bo + 48 * N_NODES);

        acc[0] = __builtin_amdgcn_wmma_f32_16x16x32_bf16(false, afrag, false, b0, (short)0, acc[0], false, false);
        acc[1] = __builtin_amdgcn_wmma_f32_16x16x32_bf16(false, afrag, false, b1, (short)0, acc[1], false, false);
        acc[2] = __builtin_amdgcn_wmma_f32_16x16x32_bf16(false, afrag, false, b2, (short)0, acc[2], false, false);
        acc[3] = __builtin_amdgcn_wmma_f32_16x16x32_bf16(false, afrag, false, b3, (short)0, acc[3], false, false);
    }

    // ---- publish partials to LDS ----
    if (hi == 0) {
        ls_m[(head * N_SEG + seg) * 16 + m] = mrun;
        ls_l[(head * N_SEG + seg) * 16 + m] = lrun;
    }
#pragma unroll
    for (int t = 0; t < 4; ++t)
#pragma unroll
        for (int r = 0; r < 8; ++r)
            ls_acc[((head * N_SEG + seg) * 16 + (8 * hi + r)) * OUT_FEAT + t * 16 + m] = acc[t][r];
    __syncthreads();

    // ---- combine segments and write output ----
    // 4 heads * 16 rows * 64 feats = 4096 outputs; 512 threads -> 8 each
#pragma unroll
    for (int i = 0; i < 8; ++i) {
        int oidx = threadIdx.x + i * 512;         // 0..4095
        int h2   = oidx >> 10;
        int r2   = (oidx >> 6) & 15;
        int f2   = oidx & 63;
        const float* mp = ls_m + (h2 * N_SEG) * 16 + r2;
        const float* lp = ls_l + (h2 * N_SEG) * 16 + r2;
        float mstar = fmaxf(fmaxf(mp[0], mp[16]), fmaxf(mp[32], mp[48]));
        float li = 0.f, v = 0.f;
#pragma unroll
        for (int sg = 0; sg < N_SEG; ++sg) {
            float sc = __builtin_amdgcn_exp2f(mp[sg * 16] - mstar);
            li += lp[sg * 16] * sc;
            v  += ls_acc[((h2 * N_SEG + sg) * 16 + r2) * OUT_FEAT + f2] * sc;
        }
        out[(size_t)(row0 + r2) * CDIM + h2 * OUT_FEAT + f2] =
            v * __builtin_amdgcn_rcpf(li) + bias[h2 * OUT_FEAT + f2];
    }
}

// ---------------------------------------------------------------------------
extern "C" void kernel_launch(void* const* d_in, const int* in_sizes, int n_in,
                              void* d_out, int out_size, void* d_ws, size_t ws_size,
                              hipStream_t stream) {
    const float* h    = (const float*)d_in[0];
    const int*   adj  = (const int*)d_in[1];
    const float* W    = (const float*)d_in[2];
    const float* a    = (const float*)d_in[3];
    const float* bias = (const float*)d_in[4];
    float* out = (float*)d_out;

    char* ws = (char*)d_ws;
    float*  wh     = (float*)ws;                                          // 4 MB fp32 Wh [node][256]
    __bf16* whT    = (__bf16*)(ws + (size_t)N_NODES * CDIM * 4);          // 2 MB bf16 Wh^T [feat][node]
    float*  leftT  = (float*)(ws + (size_t)N_NODES * CDIM * 6);           // 64 KB [head][node]
    float*  rightT = leftT + N_HEADS * N_NODES;                           // 64 KB [head][node]

    k_gemm_wh<<<(N_NODES * CDIM) / 256, 256, 0, stream>>>(h, W, wh, whT);
    k_leftright<<<(N_HEADS * N_NODES) / 256, 256, 0, stream>>>(wh, a, leftT, rightT);

    size_t lds_bytes = (size_t)(N_HEADS * N_SEG * 16 * OUT_FEAT + 2 * N_HEADS * N_SEG * 16) * sizeof(float);
    k_gat_attn<<<N_NODES / 16, 512, lds_bytes, stream>>>(adj, leftT, rightT, whT, bias, out);
}
